// SelfAttention_8478265442442
// MI455X (gfx1250) — matrix operand order repaired and checked by reading, not compiled, not run on hardware
//
#include <hip/hip_runtime.h>
#include <stdint.h>

// ---------------------------------------------------------------------------
// Self-attention forward for MI455X (gfx1250, wave32, WMMA + async-to-LDS).
// Pipeline: f32->bf16 convert, QKV GEMM (WMMA bf16), flash attention (WMMA),
// proj GEMM (WMMA bf16, f32 out). Tile staging uses double-buffered
// GLOBAL_LOAD_ASYNC_TO_LDS_B128 (ASYNCcnt) so DMA overlaps WMMA compute.
// ---------------------------------------------------------------------------

typedef __bf16 bf16;
typedef __attribute__((ext_vector_type(16))) __bf16 v16bf;
typedef __attribute__((ext_vector_type(8)))  float  v8f;
typedef __attribute__((ext_vector_type(8)))  __bf16 bf16x8;
typedef __attribute__((ext_vector_type(4)))  __bf16 bf16x4;

#define EMBED 1024
#define HEADS 16
#define BATCH 4
#define SEQ   2048
#define HS    64
#define M_ROWS (BATCH * SEQ)   // 8192

// ---------------------------------------------------------------------------
// CDNA5 async copy: 16B per lane, global -> LDS, tracked by ASYNCcnt.
// Generic pointers to LDS carry the LDS byte offset in the low 32 bits
// (aperture in the high word), so truncation gives the DS address.
// ---------------------------------------------------------------------------
__device__ __forceinline__ void async_b128(void* lds_dst, const void* gsrc) {
  asm volatile("global_load_async_to_lds_b128 %0, %1, off"
               :
               : "v"((uint32_t)(uintptr_t)lds_dst), "v"(gsrc)
               : "memory");
}
__device__ __forceinline__ void wait_async0() {
  asm volatile("s_wait_asynccnt 0x0" ::: "memory");
}

// ---------------------------------------------------------------------------
// WMMA helpers (wave32; A and B both use contiguous k = 16*half + s)
// ---------------------------------------------------------------------------
__device__ __forceinline__ v8f wmma_bf16(v16bf a, v16bf b, v8f c) {
  return __builtin_amdgcn_wmma_f32_16x16x32_bf16(
      false, a, false, b, (short)0, c, false, false);
}

// A fragment: 16x32 bf16, row-major tile, leading dim ld.
__device__ __forceinline__ v16bf frag_a(const bf16* p, int ld, int lane) {
  const int half = lane >> 4;
  const bf16* r = p + (lane & 15) * ld + half * 16;
  v16bf f;
#pragma unroll
  for (int i = 0; i < 4; ++i) {
    f[2*i]     = r[2*i];
    f[2*i+1]   = r[2*i+1];
    f[8+2*i]   = r[8 + 2*i];
    f[8+2*i+1] = r[8 + 2*i+1];
  }
  return f;
}

// B fragment from N-major storage: Bs[n][k] (B^T row-major), leading dim ld.
__device__ __forceinline__ v16bf frag_b_nk(const bf16* p, int ld, int lane) {
  const int half = lane >> 4;
  const bf16* c = p + (lane & 15) * ld + half * 16;
  v16bf f;
#pragma unroll
  for (int i = 0; i < 8; ++i) {
    f[2*i]   = c[2*i];
    f[2*i+1] = c[2*i+1];
  }
  return f;
}

// B fragment from K-major storage: Bs[k][n], leading dim ld.
__device__ __forceinline__ v16bf frag_b_kn(const bf16* p, int ld, int lane) {
  const int half = lane >> 4;
  const bf16* c = p + (lane & 15) + half * 16 * ld;
  v16bf f;
#pragma unroll
  for (int i = 0; i < 8; ++i) {
    f[2*i]   = c[(2*i) * ld];
    f[2*i+1] = c[(2*i+1) * ld];
  }
  return f;
}

// ---------------------------------------------------------------------------
// Kernel 0: f32 -> bf16 conversion (vectorized)
// ---------------------------------------------------------------------------
__global__ __launch_bounds__(256)
void cvt_f32_bf16(const float* __restrict__ in, bf16* __restrict__ out, int n4) {
  int i = blockIdx.x * blockDim.x + threadIdx.x;
  if (i < n4) {
    const float4 v = reinterpret_cast<const float4*>(in)[i];
    bf16x4 o;
    o[0] = (bf16)v.x; o[1] = (bf16)v.y; o[2] = (bf16)v.z; o[3] = (bf16)v.w;
    reinterpret_cast<bf16x4*>(out)[i] = o;
  }
}

// ---------------------------------------------------------------------------
// Kernel 1/3: GEMM  out[M,N] = A[M,K] * W[N,K]^T + bias
// Block tile 128x128, K step 32, 256 threads = 8 waves, wave tile 32x64.
// Double-buffered async-to-LDS staging.
// ---------------------------------------------------------------------------
template <bool WRITE_F32>
__global__ __launch_bounds__(256)
void gemm_bf16_nt(const bf16* __restrict__ A, const bf16* __restrict__ W,
                  const float* __restrict__ bias, void* __restrict__ out,
                  int M, int N, int K) {
  constexpr int LDA = 40;   // 32 + 8 pad (bf16 elems)
  __shared__ bf16 As[2][128 * LDA];
  __shared__ bf16 Ws[2][128 * LDA];

  const int tid  = threadIdx.x;
  const int lane = tid & 31;
  const int wave = tid >> 5;           // 0..7
  const int wm   = (wave >> 1) * 32;   // 0,32,64,96
  const int wn   = (wave & 1) * 64;    // 0,64
  const int m0   = blockIdx.y * 128;
  const int n0   = blockIdx.x * 128;

  // per-thread staging coordinates: 128x32 tile = 512 x 16B chunks
  const int r0 = (tid * 2) >> 2;             // row of chunk 0 (chunk1 same or +? no: tid*2, tid*2+1)
  // (recomputed inline below for clarity)

  auto stage = [&](int buf, int k0) {
#pragma unroll
    for (int j = 0; j < 2; ++j) {
      const int chunk = tid * 2 + j;         // 0..511
      const int row   = chunk >> 2;
      const int koff  = (chunk & 3) * 8;
      async_b128(&As[buf][row * LDA + koff],
                 A + (size_t)(m0 + row) * K + k0 + koff);
      async_b128(&Ws[buf][row * LDA + koff],
                 W + (size_t)(n0 + row) * K + k0 + koff);
    }
  };

  v8f acc[2][4] = {};

  stage(0, 0);
  wait_async0();
  __syncthreads();

  for (int k0 = 0; k0 < K; k0 += 32) {
    const int cur = (k0 >> 5) & 1;
    if (k0 + 32 < K) stage(cur ^ 1, k0 + 32);   // prefetch next tile (async)

    const v16bf af0 = frag_a(&As[cur][(wm +  0) * LDA], LDA, lane);
    const v16bf af1 = frag_a(&As[cur][(wm + 16) * LDA], LDA, lane);
#pragma unroll
    for (int j = 0; j < 4; ++j) {
      const v16bf bfj = frag_b_nk(&Ws[cur][(wn + j * 16) * LDA], LDA, lane);
      acc[0][j] = wmma_bf16(af0, bfj, acc[0][j]);
      acc[1][j] = wmma_bf16(af1, bfj, acc[1][j]);
    }

    wait_async0();       // own async copies into buf^1 done
    __syncthreads();     // everyone done computing on cur + staging
  }

  // epilogue: bias + store (C/D layout: row = half*8+e, col = lane&15)
  const int half = lane >> 4;
  const int col  = lane & 15;
#pragma unroll
  for (int i = 0; i < 2; ++i) {
#pragma unroll
    for (int j = 0; j < 4; ++j) {
      const int gn = n0 + wn + j * 16 + col;
      const float bv = bias[gn];
#pragma unroll
      for (int e = 0; e < 8; ++e) {
        const int gm = m0 + wm + i * 16 + half * 8 + e;
        const float v = acc[i][j][e] + bv;
        if (WRITE_F32)
          ((float*)out)[(size_t)gm * N + gn] = v;
        else
          ((bf16*)out)[(size_t)gm * N + gn] = (bf16)v;
      }
    }
  }
}

// ---------------------------------------------------------------------------
// Kernel 2: flash attention (causal). 128 q-rows per WG, 16 per wave.
// K/V tiles double-buffered via async-to-LDS.
// ---------------------------------------------------------------------------
__global__ __launch_bounds__(256)
void flash_attn(const bf16* __restrict__ qkv, bf16* __restrict__ y) {
  constexpr int LDT = 72;              // 64 + 8 pad
  __shared__ bf16 Qs[128 * LDT];
  __shared__ bf16 Ks[2][64 * LDT];
  __shared__ bf16 Vs[2][64 * LDT];
  __shared__ bf16 Ps[8 * 16 * LDT];    // per-wave P slab

  const int tid  = threadIdx.x;
  const int lane = tid & 31;
  const int wave = tid >> 5;
  const int half = lane >> 4;
  const int col  = lane & 15;

  const int b  = blockIdx.z;
  const int h  = blockIdx.y;
  const int q0 = blockIdx.x * 128;
  const size_t rs = 3 * EMBED;
  const bf16* qb = qkv + (size_t)b * SEQ * rs + 0 * EMBED + h * HS;
  const bf16* kb = qkv + (size_t)b * SEQ * rs + 1 * EMBED + h * HS;
  const bf16* vb = qkv + (size_t)b * SEQ * rs + 2 * EMBED + h * HS;

  auto stageKV = [&](int buf, int k0r) {
#pragma unroll
    for (int j = 0; j < 2; ++j) {
      const int chunk = tid + j * 256;   // 0..511
      const int row   = chunk >> 3;
      const int koff  = (chunk & 7) * 8;
      async_b128(&Ks[buf][row * LDT + koff],
                 kb + (size_t)(k0r + row) * rs + koff);
      async_b128(&Vs[buf][row * LDT + koff],
                 vb + (size_t)(k0r + row) * rs + koff);
    }
  };

  // kick off K/V tile 0 (async) and stage Q tile 128x64 (plain)
  stageKV(0, 0);
#pragma unroll
  for (int j = 0; j < 4; ++j) {
    const int chunk = tid + j * 256;     // 0..1023
    const int row   = chunk >> 3;
    const int koff  = (chunk & 7) * 8;
    *(bf16x8*)(Qs + row * LDT + koff) =
        *(const bf16x8*)(qb + (size_t)(q0 + row) * rs + koff);
  }
  wait_async0();
  __syncthreads();

  const v16bf qf0 = frag_a(Qs + wave * 16 * LDT +  0, LDT, lane);
  const v16bf qf1 = frag_a(Qs + wave * 16 * LDT + 32, LDT, lane);

  const int rowg0 = q0 + wave * 16 + half * 8;
  const int wave_rmax = q0 + wave * 16 + 15;

  v8f   o[4] = {};
  float m[8], l[8];
#pragma unroll
  for (int i = 0; i < 8; ++i) { m[i] = -3.0e38f; l[i] = 0.0f; }

  const int jmax = (q0 + 127) >> 6;
  for (int jt = 0; jt <= jmax; ++jt) {
    const int k0r = jt * 64;
    const int cur = jt & 1;
    if (jt < jmax) stageKV(cur ^ 1, k0r + 64);   // async prefetch next tile

    if (k0r <= wave_rmax) {            // wave-uniform: EXEC stays all-1s
      // S = Q * K^T
      v8f s[4];
#pragma unroll
      for (int n = 0; n < 4; ++n) {
        v16bf kf0 = frag_b_nk(&Ks[cur][(n * 16) * LDT +  0], LDT, lane);
        v16bf kf1 = frag_b_nk(&Ks[cur][(n * 16) * LDT + 32], LDT, lane);
        v8f t = {};
        t = wmma_bf16(qf0, kf0, t);
        t = wmma_bf16(qf1, kf1, t);
        s[n] = t;
      }
      const float scale = 0.125f;      // 1/sqrt(64)
#pragma unroll
      for (int n = 0; n < 4; ++n) {
        const int c = k0r + n * 16 + col;
#pragma unroll
        for (int i = 0; i < 8; ++i) {
          const float v = s[n][i] * scale;
          s[n][i] = (c <= rowg0 + i) ? v : -3.0e38f;
        }
      }
      float mnew[8], alpha[8];
#pragma unroll
      for (int i = 0; i < 8; ++i) {
        float v = fmaxf(fmaxf(s[0][i], s[1][i]), fmaxf(s[2][i], s[3][i]));
        v = fmaxf(v, __shfl_xor(v, 1, 32));
        v = fmaxf(v, __shfl_xor(v, 2, 32));
        v = fmaxf(v, __shfl_xor(v, 4, 32));
        v = fmaxf(v, __shfl_xor(v, 8, 32));
        mnew[i]  = fmaxf(m[i], v);
        alpha[i] = __expf(m[i] - mnew[i]);
        m[i]     = mnew[i];
        l[i]    *= alpha[i];
      }
#pragma unroll
      for (int n = 0; n < 4; ++n)
#pragma unroll
        for (int i = 0; i < 8; ++i) o[n][i] *= alpha[i];
#pragma unroll
      for (int n = 0; n < 4; ++n)
#pragma unroll
        for (int i = 0; i < 8; ++i) s[n][i] = __expf(s[n][i] - mnew[i]);
#pragma unroll
      for (int i = 0; i < 8; ++i) {
        float v = s[0][i] + s[1][i] + s[2][i] + s[3][i];
        v += __shfl_xor(v, 1, 32);
        v += __shfl_xor(v, 2, 32);
        v += __shfl_xor(v, 4, 32);
        v += __shfl_xor(v, 8, 32);
        l[i] += v;
      }
      // re-fragment P as A operand through wave-private LDS slab
      bf16* pw = Ps + wave * 16 * LDT;
#pragma unroll
      for (int n = 0; n < 4; ++n)
#pragma unroll
        for (int i = 0; i < 8; ++i)
          pw[(half * 8 + i) * LDT + n * 16 + col] = (bf16)s[n][i];
      const v16bf pf0 = frag_a(pw +  0, LDT, lane);
      const v16bf pf1 = frag_a(pw + 32, LDT, lane);
      // O += P * V
#pragma unroll
      for (int n = 0; n < 4; ++n) {
        v16bf vf0 = frag_b_kn(&Vs[cur][n * 16], LDT, lane);
        v16bf vf1 = frag_b_kn(&Vs[cur][32 * LDT + n * 16], LDT, lane);
        o[n] = wmma_bf16(pf0, vf0, o[n]);
        o[n] = wmma_bf16(pf1, vf1, o[n]);
      }
    }

    wait_async0();
    __syncthreads();
  }

  float linv[8];
#pragma unroll
  for (int i = 0; i < 8; ++i) linv[i] = 1.0f / l[i];
#pragma unroll
  for (int n = 0; n < 4; ++n) {
#pragma unroll
    for (int i = 0; i < 8; ++i) {
      const int t = rowg0 + i;
      const float v = o[n][i] * linv[i];
      y[((size_t)(b * SEQ + t)) * EMBED + h * HS + n * 16 + col] = (bf16)v;
    }
  }
}

// ---------------------------------------------------------------------------
// Host launcher
// ---------------------------------------------------------------------------
extern "C" void kernel_launch(void* const* d_in, const int* in_sizes, int n_in,
                              void* d_out, int out_size, void* d_ws, size_t ws_size,
                              hipStream_t stream) {
  (void)in_sizes; (void)n_in; (void)out_size; (void)ws_size;
  const float* x      = (const float*)d_in[0];
  const float* w_qkv  = (const float*)d_in[1];
  const float* b_qkv  = (const float*)d_in[2];
  const float* w_proj = (const float*)d_in[3];
  const float* b_proj = (const float*)d_in[4];

  const int n_x     = M_ROWS * EMBED;
  const int n_wqkv  = 3 * EMBED * EMBED;
  const int n_wproj = EMBED * EMBED;

  char* ws = (char*)d_ws;
  bf16* xb    = (bf16*)(ws);                                  // 16 MiB
  bf16* wqkvb = (bf16*)(ws + (size_t)16  * 1024 * 1024);      //  6 MiB
  bf16* wprjb = (bf16*)(ws + (size_t)22  * 1024 * 1024);      //  2 MiB
  bf16* qkvb  = (bf16*)(ws + (size_t)24  * 1024 * 1024);      // 48 MiB
  bf16* yb    = (bf16*)(ws + (size_t)72  * 1024 * 1024);      // 16 MiB

  cvt_f32_bf16<<<n_x     / (256 * 4), 256, 0, stream>>>(x,      xb,    n_x / 4);
  cvt_f32_bf16<<<n_wqkv  / (256 * 4), 256, 0, stream>>>(w_qkv,  wqkvb, n_wqkv / 4);
  cvt_f32_bf16<<<n_wproj / (256 * 4), 256, 0, stream>>>(w_proj, wprjb, n_wproj / 4);

  gemm_bf16_nt<false><<<dim3(3 * EMBED / 128, M_ROWS / 128), 256, 0, stream>>>(
      xb, wqkvb, b_qkv, qkvb, M_ROWS, 3 * EMBED, EMBED);

  flash_attn<<<dim3(SEQ / 128, HEADS, BATCH), 256, 0, stream>>>(qkvb, yb);

  gemm_bf16_nt<true><<<dim3(EMBED / 128, M_ROWS / 128), 256, 0, stream>>>(
      yb, wprjb, b_proj, d_out, M_ROWS, EMBED, EMBED);
}
